// KATF_41489384079629
// MI455X (gfx1250) — matrix operand order repaired and checked by reading, not compiled, not run on hardware
//
#include <hip/hip_runtime.h>

// ---------------------------------------------------------------------------
// MI455X (gfx1250) implementation.
//  - All GEMMs (LSTM gates fused [x_t,h]@[Wih;Whh]^T, kn/attention MLPs) use
//    v_wmma_f32_16x16x32_bf16 (wave32 WMMA, f32 accumulate).
//  - Tile staging uses gfx1250 async DMA: global_load_async_to_lds_b128 with
//    double-buffered LDS; the async fill of K-tile i+1 overlaps the WMMA
//    burst on tile i, with one s_wait_asynccnt + one barrier per K-step.
//  - Per-timestep biLSTM step = 1 WMMA GEMM launch (gridDim.z = 2 directions)
//    + 1 elementwise cell kernel. 600 sequential steps (3 layers x 200); per-step working set
//    (~1.5MB Wcat + 256KB h) is L2-resident, so step latency is WMMA+LDS
//    bound -> async overlap is the right lever.
//  - Conv stack / scans are f32 VALU (bandwidth- and FLOP-trivial).
// ---------------------------------------------------------------------------

#ifndef USE_ASYNC_STAGING
#define USE_ASYNC_STAGING 1
#endif

typedef __bf16 bf16;
typedef __attribute__((ext_vector_type(16))) __bf16 v16bf;
typedef __attribute__((ext_vector_type(8)))  __bf16 v8bf;
typedef __attribute__((ext_vector_type(8)))  float  v8f;

#define EPSBN 1e-5f
#define HIDN  256

__device__ __forceinline__ float sigf(float x){ return 1.f/(1.f + __expf(-x)); }

#if USE_ASYNC_STAGING
typedef __attribute__((address_space(3))) const char lds_cchar;
__device__ __forceinline__ unsigned lds_addr32(const void* p){
  return (unsigned)(unsigned long long)(lds_cchar*)p;
}
// 16B DMA: global -> LDS, tracked by ASYNCcnt (no VGPR staging).
__device__ __forceinline__ void async_copy_b128(unsigned lds_off, const void* gaddr){
  asm volatile("global_load_async_to_lds_b128 %0, %1, off"
               :: "v"(lds_off), "v"(gaddr) : "memory");
}
__device__ __forceinline__ void wait_async0(){
  asm volatile("s_wait_asynccnt 0" ::: "memory");
}
#endif

// ---------------------------- weight preparation ---------------------------
__global__ void pack_lstm_w(const float* __restrict__ wih, const float* __restrict__ whh,
                            const float* __restrict__ bih, const float* __restrict__ bhh,
                            bf16* __restrict__ Wcat, float* __restrict__ bias, int I){
  int idx = blockIdx.x*256 + threadIdx.x;
  int KK = I + HIDN;
  int n = idx / KK, k = idx % KK;
  Wcat[idx] = (bf16)((k < I) ? wih[(long)n*I + k] : whh[(long)n*HIDN + (k - I)]);
  if (idx < 4*HIDN) bias[idx] = bih[idx] + bhh[idx];
}

__global__ void cvt_bf16_k(const float* __restrict__ s, bf16* __restrict__ d, long n){
  long i = (long)blockIdx.x*256 + threadIdx.x;
  if (i < n) d[i] = (bf16)s[i];
}

// ------------------------------- conv + BN + relu --------------------------
__global__ void conv_bn_relu(const float* __restrict__ x, const float* __restrict__ w,
                             const float* __restrict__ bc, const float* __restrict__ g,
                             const float* __restrict__ be, const float* __restrict__ mm,
                             const float* __restrict__ vv, float* __restrict__ y,
                             int Cin, int Lin, int Cout, int Lout, int Kk, int stride, int pad){
  long idx = (long)blockIdx.x*256 + threadIdx.x;
  int t = (int)(idx % Lout);
  int o = (int)((idx / Lout) % Cout);
  int b = (int)(idx / ((long)Lout*Cout));
  int base = t*stride - pad;
  int k0 = base < 0 ? -base : 0;
  int k1 = (base + Kk > Lin) ? (Lin - base) : Kk;
  float acc = 0.f;
  for (int i=0;i<Cin;i++){
    const float* xr = x + ((long)b*Cin + i)*Lin + base;
    const float* wr = w + ((long)o*Cin + i)*Kk;
    for (int k=k0;k<k1;k++) acc += xr[k]*wr[k];
  }
  float scale = g[o]*rsqrtf(vv[o]+EPSBN);
  float v = (acc + bc[o] - mm[o])*scale + be[o];
  y[idx] = fmaxf(v, 0.f);
}

// (B,C,T) f32 -> (B,T,C) bf16
__global__ void nct_to_ntc_bf16(const float* __restrict__ src, bf16* __restrict__ dst,
                                int C, int T){
  long idx = (long)blockIdx.x*256 + threadIdx.x;  // (b*T+t)*C + c
  int c = (int)(idx % C);
  long bt = idx / C;
  int t = (int)(bt % T);
  long b = bt / T;
  dst[idx] = (bf16)src[(b*C + c)*(long)T + t];
}

// -------------------------- WMMA bf16 GEMM core ----------------------------
// C(M,N) = act(A(M,K) @ B(N,K)^T + bias).  A may be split at K1 (A1|A2) for
// the fused LSTM [x_t, h] input.  Block tile 128x128, 8 waves (4M x 2N),
// wave tile 32x64 = 2x4 v_wmma_f32_16x16x32_bf16 accumulators.
// All problem dims are exact multiples of the tiles (guaranteed by launches).
template<int ACT, bool OUT_BF16>
__device__ __forceinline__ void gemm_core(
    const bf16* A1, long lda1, const bf16* A2, long lda2, int K1,
    const bf16* __restrict__ Bw, long ldb,
    const float* __restrict__ bias, void* __restrict__ Cp, long ldc,
    int K, int bm, int bn)
{
  const int tid  = threadIdx.x;
  const int lane = tid & 31;
  const int wave = tid >> 5;
  const int wm = (wave >> 1) * 32;   // wave row offset in tile
  const int wn = (wave & 1) * 64;    // wave col offset in tile

  v8f acc[2][4];
#pragma unroll
  for (int i=0;i<2;i++)
#pragma unroll
    for (int j=0;j<4;j++)
#pragma unroll
      for (int r=0;r<8;r++) acc[i][j][r] = 0.f;

  const int lrow = tid >> 1;           // 0..127
  const int lcc  = (tid & 1) * 16;     // 0 or 16
  const int mlo  = lane & 15;
  const int khiA = (lane >> 4) * 8;
  const int kbB  = (lane >> 4) * 8;

#if USE_ASYNC_STAGING
  // Double-buffered LDS, filled by async DMA (ASYNCcnt).  Stride 40 bf16
  // keeps the b128 fragment loads bank-conflict free.
  __shared__ bf16 As[2][128*40];
  __shared__ bf16 Bs[2][128*40];

  auto issue_stage = [&](int k0, int buf){
    const int kc = k0 + lcc;           // multiple of 16; K1 is too -> no straddle
    const long grow = (long)bm*128 + lrow;
    const bf16* asrc = (kc < K1) ? (A1 + grow*lda1 + kc)
                                 : (A2 + grow*lda2 + (kc - K1));
    const bf16* bsrc = Bw + ((long)bn*128 + lrow)*ldb + kc;
    unsigned la = lds_addr32(&As[buf][lrow*40 + lcc]);
    unsigned lb = lds_addr32(&Bs[buf][lrow*40 + lcc]);
    async_copy_b128(la,      asrc);
    async_copy_b128(la + 16, asrc + 8);
    async_copy_b128(lb,      bsrc);
    async_copy_b128(lb + 16, bsrc + 8);
  };

  const int nIter = K >> 5;
  issue_stage(0, 0);
  for (int it = 0; it < nIter; ++it) {
    const int cur = it & 1;
    wait_async0();                     // tile[cur] landed in LDS (this wave)
    __syncthreads();                   // ... and everyone else's portion too
    if (it + 1 < nIter) issue_stage((it + 1) << 5, 1 - cur);  // overlap fill
    const bf16* Ac = As[cur];
    const bf16* Bc = Bs[cur];
#else
  __shared__ bf16 As1[128*40];
  __shared__ bf16 Bs1[128*40];
  const int nIter = K >> 5;
  for (int it = 0; it < nIter; ++it) {
    const int kc = (it << 5) + lcc;
    const long grow = (long)bm*128 + lrow;
    const bf16* asrc = (kc < K1) ? (A1 + grow*lda1 + kc)
                                 : (A2 + grow*lda2 + (kc - K1));
    v8bf a0 = *(const v8bf*)asrc;
    v8bf a1 = *(const v8bf*)(asrc + 8);
    const bf16* bsrc = Bw + ((long)bn*128 + lrow)*ldb + kc;
    v8bf b0 = *(const v8bf*)bsrc;
    v8bf b1 = *(const v8bf*)(bsrc + 8);
    __syncthreads();
    *(v8bf*)(&As1[lrow*40 + lcc    ]) = a0;
    *(v8bf*)(&As1[lrow*40 + lcc + 8]) = a1;
    *(v8bf*)(&Bs1[lrow*40 + lcc    ]) = b0;
    *(v8bf*)(&Bs1[lrow*40 + lcc + 8]) = b1;
    __syncthreads();
    const bf16* Ac = As1;
    const bf16* Bc = Bs1;
#endif

    // ---- LDS -> fragments (ISA 16-bit A 16x32 / B 32x16 lane striping) ----
    v16bf af[2];
#pragma unroll
    for (int i=0;i<2;i++){
      const bf16* p = &Ac[(wm + i*16 + mlo)*40 + khiA];
      v8bf lo = *(const v8bf*)p;
      v8bf hi = *(const v8bf*)(p + 16);
#pragma unroll
      for (int e=0;e<8;e++){ af[i][e] = lo[e]; af[i][e+8] = hi[e]; }
    }
    v16bf bfr[4];
#pragma unroll
    for (int j=0;j<4;j++){
      const bf16* p = &Bc[(wn + j*16 + mlo)*40 + kbB];
      v8bf lo = *(const v8bf*)p;
      v8bf hi = *(const v8bf*)(p + 16);
#pragma unroll
      for (int e=0;e<8;e++){ bfr[j][e] = lo[e]; bfr[j][e+8] = hi[e]; }
    }

#pragma unroll
    for (int i=0;i<2;i++)
#pragma unroll
      for (int j=0;j<4;j++)
        acc[i][j] = __builtin_amdgcn_wmma_f32_16x16x32_bf16(
            false, af[i], false, bfr[j], (short)0, acc[i][j], false, false);
  }

  // ---- epilogue: bias + activation, f32 or bf16 store ----
  const int mhi = (lane >> 4) * 8;
#pragma unroll
  for (int i=0;i<2;i++)
#pragma unroll
    for (int j=0;j<4;j++){
      long gn = (long)bn*128 + wn + j*16 + mlo;
      float bv = bias[gn];
#pragma unroll
      for (int r=0;r<8;r++){
        long gm = (long)bm*128 + wm + i*16 + mhi + r;
        float v = acc[i][j][r] + bv;
        if (ACT == 1) v = fmaxf(v, 0.f);
        else if (ACT == 2) v = tanhf(v);
        if (OUT_BF16) ((bf16*)Cp)[gm*ldc + gn] = (bf16)v;
        else          ((float*)Cp)[gm*ldc + gn] = v;
      }
    }
}

template<int ACT, bool OUT_BF16>
__global__ void __launch_bounds__(256)
gemm_ba(const bf16* A, long lda, const bf16* __restrict__ Bw, long ldb,
        const float* __restrict__ bias, void* __restrict__ C, long ldc, int K){
  gemm_core<ACT,OUT_BF16>(A, lda, A, lda, K, Bw, ldb, bias, C, ldc, K,
                          blockIdx.y, blockIdx.x);
}

// Fused biLSTM gate GEMM: gates[dir] = [x_tpos, h_dir] @ Wcat_dir^T + bias_dir
__global__ void __launch_bounds__(256)
lstm_gemm_step(const bf16* __restrict__ X, int I, int T, int tf, int tb,
               const bf16* __restrict__ W0, const bf16* __restrict__ W1,
               const float* __restrict__ bias, const bf16* __restrict__ h,
               float* __restrict__ gates){
  int dir = blockIdx.z;
  const bf16* A1 = X + (long)(dir ? tb : tf) * I;     // row stride T*I selects batch
  gemm_core<0,false>(A1, (long)T*I,
                     h + (long)dir*512*HIDN, HIDN, I,
                     dir ? W1 : W0, I + HIDN,
                     bias + dir*4*HIDN,
                     gates + (long)dir*512*4*HIDN, 4*HIDN,
                     I + HIDN, blockIdx.y, blockIdx.x);
}

// ------------------------------ LSTM cell ----------------------------------
__global__ void lstm_cell_k(const float* __restrict__ gates, bf16* __restrict__ h,
                            float* __restrict__ c, bf16* __restrict__ Xout,
                            int T, int tf, int tb){
  int idx = blockIdx.x*256 + threadIdx.x;       // 2*512*256 threads
  int u   = idx & 255;
  int b   = (idx >> 8) & 511;
  int dir = idx >> 17;
  const float* g = gates + ((long)dir*512 + b)*1024;
  float gi = g[u], gf = g[256+u], gg = g[512+u], go = g[768+u];
  long si = ((long)dir*512 + b)*256 + u;
  float cc = c[si];
  cc = sigf(gf)*cc + sigf(gi)*tanhf(gg);
  float hh = sigf(go)*tanhf(cc);
  c[si] = cc;
  h[si] = (bf16)hh;
  int tpos = dir ? tb : tf;
  Xout[((long)b*T + tpos)*512 + dir*256 + u] = (bf16)hh;
}

// ---------------------- Kalman gain + scan ---------------------------------
__global__ void kn_gain(const bf16* __restrict__ hid, const float* __restrict__ w2,
                        const float* __restrict__ b2, float* __restrict__ Kg){
  long row = (long)blockIdx.x*256 + threadIdx.x;  // B*T rows
  const bf16* hr = hid + row*1024;
  float a0=0.f, a1=0.f, a2=0.f;
  for (int k=0;k<1024;k++){
    float hv = (float)hr[k];
    a0 += hv*w2[k]; a1 += hv*w2[1024+k]; a2 += hv*w2[2048+k];
  }
  a0 += b2[0]; a1 += b2[1]; a2 += b2[2];
  float alpha = sigf(a0);
  float Q = __expf(a1), R = __expf(a2);
  Kg[row] = alpha * (Q / fmaxf(Q + R, 1e-6f));
}

__global__ void kalman_scan(const bf16* __restrict__ lstm, const float* __restrict__ Kg,
                            bf16* __restrict__ kal, int T){
  int idx = blockIdx.x*256 + threadIdx.x;  // 512*512
  int d = idx & 511;
  int b = idx >> 9;
  float state = (float)lstm[((long)b*T)*512 + d];
  kal[((long)b*T)*512 + d] = (bf16)state;   // t=0: K0*(m0-m0)=0
  for (int t=1;t<T;t++){
    float m = (float)lstm[((long)b*T + t)*512 + d];
    state += Kg[(long)b*T + t]*(m - state);
    kal[((long)b*T + t)*512 + d] = (bf16)state;
  }
}

// ---------------------- attention scores / softmax / ctx -------------------
__global__ void attn_scores(const bf16* __restrict__ h2, const float* __restrict__ w,
                            const float* __restrict__ bb, float* __restrict__ sc){
  long row = (long)blockIdx.x*256 + threadIdx.x;
  const bf16* hr = h2 + row*256;
  float acc = 0.f;
  for (int k=0;k<256;k++) acc += (float)hr[k]*w[k];
  sc[row] = acc + bb[0];
}

__global__ void attn_softmax_ctx(const float* __restrict__ sc, const bf16* __restrict__ kal,
                                 float* __restrict__ ctx, int T){
  int b = blockIdx.x, tid = threadIdx.x;
  __shared__ float w[256];
  __shared__ float red[256];
  float s = (tid < T) ? sc[(long)b*T + tid] : -1e30f;
  red[tid] = s; __syncthreads();
  for (int off=128; off>0; off>>=1){ if (tid<off) red[tid]=fmaxf(red[tid],red[tid+off]); __syncthreads(); }
  float mx = red[0]; __syncthreads();
  float e = (tid < T) ? __expf(s - mx) : 0.f;
  red[tid] = e; __syncthreads();
  for (int off=128; off>0; off>>=1){ if (tid<off) red[tid]+=red[tid+off]; __syncthreads(); }
  float inv = 1.f/red[0];
  w[tid] = e*inv;
  __syncthreads();
  for (int d=tid; d<512; d+=256){
    float acc = 0.f;
    for (int t=0;t<T;t++) acc += w[t]*(float)kal[((long)b*T + t)*512 + d];
    ctx[(long)b*512 + d] = acc;
  }
}

// ---------------------------- classifier -----------------------------------
__global__ void cl1_k(const float* __restrict__ ctx, const float* __restrict__ w1,
                      const float* __restrict__ b1, const float* __restrict__ g,
                      const float* __restrict__ be, const float* __restrict__ mm,
                      const float* __restrict__ vv, float* __restrict__ clh){
  int idx = blockIdx.x*256 + threadIdx.x;   // 512*256
  int j = idx & 255, b = idx >> 8;
  const float* cr = ctx + (long)b*512;
  const float* wr = w1 + (long)j*512;
  float acc = 0.f;
  for (int k=0;k<512;k++) acc += cr[k]*wr[k];
  float h = fmaxf(acc + b1[j], 0.f);
  clh[idx] = (h - mm[j])*(g[j]*rsqrtf(vv[j]+EPSBN)) + be[j];
}

__global__ void cl2_k(const float* __restrict__ clh, const float* __restrict__ w2,
                      const float* __restrict__ b2, float* __restrict__ out){
  int idx = blockIdx.x*256 + threadIdx.x;   // 512*4
  int c = idx & 3, b = idx >> 2;
  const float* hr = clh + (long)b*256;
  const float* wr = w2 + (long)c*256;
  float acc = 0.f;
  for (int k=0;k<256;k++) acc += hr[k]*wr[k];
  out[idx] = acc + b2[c];
}

// ===========================================================================
extern "C" void kernel_launch(void* const* d_in, const int* in_sizes, int n_in,
                              void* d_out, int out_size, void* d_ws, size_t ws_size,
                              hipStream_t stream){
  (void)in_sizes; (void)n_in; (void)out_size; (void)ws_size;
  const float* x = (const float*)d_in[0];
  // conv params: c{1..4}: w,b,g,be,m,v at indices 1..24
  auto cw = [&](int c, int j){ return (const float*)d_in[1 + c*6 + j]; };

  // ---- workspace layout (cursor, 256B aligned); total ~460 MB ----
  char* W = (char*)d_ws; size_t off = 0;
  auto take = [&](size_t bytes)->void*{
    void* p = W + off; off = (off + bytes + 255) & ~(size_t)255; return p; };
  bf16*  WL    = (bf16*) take(4456448ull*2);   // packed [Wih;Whh] bf16, 6 slots
  float* biasL = (float*)take(6144ull*4);      // bih+bhh, 6 x 1024
  bf16*  knw1  = (bf16*) take(524288ull*2);    // kn_w1 bf16 (1024x512)
  bf16*  atw1  = (bf16*) take(131072ull*2);    // at_w1 bf16 (256x512)
  bf16*  hbuf  = (bf16*) take(262144ull*2);    // h state, 2 dirs x 512 x 256
  float* cbuf  = (float*)take(262144ull*4);    // c state
  float* gates = (float*)take(1048576ull*4);   // 2 x 512 x 1024
  float* Kg    = (float*)take(102400ull*4);
  float* sco   = (float*)take(102400ull*4);
  float* ctx   = (float*)take(262144ull*4);
  float* clh   = (float*)take(131072ull*4);
  char*  bufA  = (char*) take(120000000ull);   // h1 / Xin / kal
  char*  bufB  = (char*) take(210000000ull);   // h2 / h4 / kn hidden
  char*  bufC  = (char*) take(110500000ull);   // h3 / Xout / lstm_out / hidden2

  // ---- pack LSTM weights (bf16, [Wih;Whh] concat along K) ----
  const size_t wlOff[6] = {0, 655360, 1310720, 2097152, 2883584, 3670016};
  for (int l=0;l<3;l++) for (int d=0;d<2;d++){
    int I = l ? 512 : 384;
    int slot = l*2 + d;
    int total = 1024*(I + 256);
    pack_lstm_w<<<total/256, 256, 0, stream>>>(
        (const float*)d_in[25 + slot*4 + 0], (const float*)d_in[25 + slot*4 + 1],
        (const float*)d_in[25 + slot*4 + 2], (const float*)d_in[25 + slot*4 + 3],
        WL + wlOff[slot], biasL + slot*1024, I);
  }
  cvt_bf16_k<<<524288/256, 256, 0, stream>>>((const float*)d_in[49], knw1, 524288);
  cvt_bf16_k<<<131072/256, 256, 0, stream>>>((const float*)d_in[53], atw1, 131072);

  // ---- conv stack (f32 VALU; L2-resident, <13% of FLOPs) ----
  float* h1 = (float*)bufA;  float* h2 = (float*)bufB;
  float* h3 = (float*)bufC;  float* h4 = (float*)bufB;
  conv_bn_relu<<<(512L*48*1200)/256, 256, 0, stream>>>(x,
      cw(0,0),cw(0,1),cw(0,2),cw(0,3),cw(0,4),cw(0,5), h1, 6,6000,48,1200,25,5,12);
  conv_bn_relu<<<(512L*96*400)/256, 256, 0, stream>>>(h1,
      cw(1,0),cw(1,1),cw(1,2),cw(1,3),cw(1,4),cw(1,5), h2, 48,1200,96,400,15,3,7);
  conv_bn_relu<<<(512L*192*200)/256, 256, 0, stream>>>(h2,
      cw(2,0),cw(2,1),cw(2,2),cw(2,3),cw(2,4),cw(2,5), h3, 96,400,192,200,7,2,3);
  conv_bn_relu<<<(512L*384*200)/256, 256, 0, stream>>>(h3,
      cw(3,0),cw(3,1),cw(3,2),cw(3,3),cw(3,4),cw(3,5), h4, 192,200,384,200,5,1,2);

  // (B,384,200) f32 -> (B,200,384) bf16
  nct_to_ntc_bf16<<<(512L*384*200)/256, 256, 0, stream>>>(h4, (bf16*)bufA, 384, 200);

  // ---- 3-layer biLSTM: 200 steps x (fused WMMA gate GEMM + cell) ----
  bf16* Xin  = (bf16*)bufA;
  bf16* Xout = (bf16*)bufC;
  for (int l=0;l<3;l++){
    int I = l ? 512 : 384;
    const bf16* W0 = WL + wlOff[l*2 + 0];
    const bf16* W1 = WL + wlOff[l*2 + 1];
    hipMemsetAsync(hbuf, 0, 262144ull*2, stream);
    hipMemsetAsync(cbuf, 0, 262144ull*4, stream);
    for (int t=0;t<200;t++){
      lstm_gemm_step<<<dim3(8,4,2), 256, 0, stream>>>(
          Xin, I, 200, t, 199-t, W0, W1, biasL + l*2*1024, hbuf, gates);
      lstm_cell_k<<<1024, 256, 0, stream>>>(gates, hbuf, cbuf, Xout, 200, t, 199-t);
    }
    bf16* tmp = Xin; Xin = Xout; Xout = tmp;
  }
  bf16* lstm_out = Xin;   // == bufC after 3 swaps

  // ---- Kalman head: hidden = relu(lstm_out @ kn_w1^T + b1)  (WMMA) ----
  gemm_ba<1,true><<<dim3(8,800), 256, 0, stream>>>(
      lstm_out, 512, knw1, 512, (const float*)d_in[50], bufB, 1024, 512);
  kn_gain<<<400, 256, 0, stream>>>((const bf16*)bufB,
      (const float*)d_in[51], (const float*)d_in[52], Kg);
  kalman_scan<<<1024, 256, 0, stream>>>(lstm_out, Kg, (bf16*)bufA, 200);

  // ---- attention: hidden2 = tanh(kal @ at_w1^T + b1)  (WMMA) ----
  gemm_ba<2,true><<<dim3(2,800), 256, 0, stream>>>(
      (const bf16*)bufA, 512, atw1, 512, (const float*)d_in[54], bufC, 256, 512);
  attn_scores<<<400, 256, 0, stream>>>((const bf16*)bufC,
      (const float*)d_in[55], (const float*)d_in[56], sco);
  attn_softmax_ctx<<<512, 256, 0, stream>>>(sco, (const bf16*)bufA, ctx, 200);

  // ---- classifier ----
  cl1_k<<<512, 256, 0, stream>>>(ctx, (const float*)d_in[57], (const float*)d_in[58],
      (const float*)d_in[59], (const float*)d_in[60], (const float*)d_in[61],
      (const float*)d_in[62], clh);
  cl2_k<<<8, 256, 0, stream>>>(clh, (const float*)d_in[63], (const float*)d_in[64],
      (float*)d_out);
}
